// SpatialTransformer_51024211476931
// MI455X (gfx1250) — compile-verified, matrix operand order checked by reading, not run on hardware
//
#include <hip/hip_runtime.h>
#include <hip/hip_bf16.h>

typedef __attribute__((ext_vector_type(16))) __bf16 v16bf;
typedef __attribute__((ext_vector_type(8)))  __bf16 bf16x8;
typedef __attribute__((ext_vector_type(4)))  __bf16 bf16x4;
typedef __attribute__((ext_vector_type(4)))  float  f32x4;
typedef __attribute__((ext_vector_type(8)))  float  v8f;

#define B_   4
#define SQ_  2048
#define SKV_ 2048
#define C_   1024

enum { OUT_BF16 = 1, OUT_TRANS = 2, ADD_RES = 4 };

// --- gfx1250 async global->LDS copy (ASYNCcnt-tracked, no VGPR staging) ---
__device__ __forceinline__ void async_copy_b128(const __bf16* gsrc, __bf16* ldst) {
#if defined(__gfx1250__)
  unsigned lds = (unsigned)(uintptr_t)ldst;          // low 32 bits = LDS offset
  unsigned long long g = (unsigned long long)(uintptr_t)gsrc;
  asm volatile("global_load_async_to_lds_b128 %0, %1, off"
               :: "v"(lds), "v"(g) : "memory");
#else
  *(bf16x8*)ldst = *(const bf16x8*)gsrc;
#endif
}
__device__ __forceinline__ void async_wait_all() {
#if defined(__gfx1250__)
  asm volatile("s_wait_asynccnt 0x0" ::: "memory");
#endif
}

// Load one 16x32 bf16 WMMA operand fragment (A-layout == B-layout for
// row-gathered operands on gfx1250):
//   lane<16 : row = lane,    K = {0..7} then {16..23}
//   lane>=16: row = lane-16, K = {8..15} then {24..31}
__device__ __forceinline__ v16bf load_frag(const __bf16* base, int ld) {
  const int lane = threadIdx.x & 31;
  const int r    = lane & 15;
  const int hi   = lane >> 4;
  const __bf16* p = base + (size_t)r * ld + hi * 8;
  bf16x8 lo = *(const bf16x8*)(p);
  bf16x8 hh = *(const bf16x8*)(p + 16);
  v16bf v;
#pragma unroll
  for (int i = 0; i < 8; ++i) { v[i] = lo[i]; v[i + 8] = hh[i]; }
  return v;
}

// Generic GEMM over row sets: Y[m][n] = sum_k A[m][k] * Bm[n][k]  (+bias,+resid)
// Workgroup tile 256x64 (8 waves), wave tile 32x64: 8 WMMAs / k-step.
// Double-buffered LDS tiles filled by async global->LDS copies.
__global__ void __launch_bounds__(256) gemm_rr(
    const __bf16* __restrict__ A, const __bf16* __restrict__ Bm,
    const float* __restrict__ bias, const float* __restrict__ resid,
    float* __restrict__ outF, __bf16* __restrict__ outH,
    int M, int N, int Kd, int ldT, int flags)
{
  __shared__ __bf16 sA[2][256 * 32];   // 2 x 16 KiB
  __shared__ __bf16 sB[2][64 * 32];    // 2 x  4 KiB
  const int tid  = threadIdx.x;
  const int lane = tid & 31;
  const int wave = tid >> 5;
  const int mBase = blockIdx.y * 256;
  const int nBase = blockIdx.x * 64;

  auto load_tiles = [&](int buf, int kb) {
#pragma unroll
    for (int l = 0; l < 4; ++l) {               // A: 256x32 = 1024 x b128
      int e = (tid + l * 256) * 8;
      int row = e >> 5, col = e & 31;
      async_copy_b128(&A[(size_t)(mBase + row) * Kd + kb + col],
                      &sA[buf][row * 32 + col]);
    }
    {                                            // B: 64x32 = 256 x b128
      int e = tid * 8;
      int row = e >> 5, col = e & 31;
      async_copy_b128(&Bm[(size_t)(nBase + row) * Kd + kb + col],
                      &sB[buf][row * 32 + col]);
    }
  };

  v8f acc[2][4];
#pragma unroll
  for (int mi = 0; mi < 2; ++mi)
#pragma unroll
    for (int t = 0; t < 4; ++t)
#pragma unroll
      for (int i = 0; i < 8; ++i) acc[mi][t][i] = 0.f;

  const int nk = Kd / 32;
  load_tiles(0, 0);
  async_wait_all();
  __syncthreads();

  for (int it = 0; it < nk; ++it) {
    const int cur = it & 1;
    if (it + 1 < nk) load_tiles(cur ^ 1, (it + 1) * 32);  // prefetch next tile

    v16bf af0 = load_frag(&sA[cur][(wave * 32) * 32], 32);
    v16bf af1 = load_frag(&sA[cur][(wave * 32 + 16) * 32], 32);
#pragma unroll
    for (int t = 0; t < 4; ++t) {
      v16bf bfr = load_frag(&sB[cur][t * 16 * 32], 32);
      acc[0][t] = __builtin_amdgcn_wmma_f32_16x16x32_bf16(
          false, af0, false, bfr, (short)0, acc[0][t], false, false);
      acc[1][t] = __builtin_amdgcn_wmma_f32_16x16x32_bf16(
          false, af1, false, bfr, (short)0, acc[1][t], false, false);
    }
    async_wait_all();
    __syncthreads();
  }

  // D layout: VGPR r -> row (r + 8*(lane>=16)), lane&15 -> col within tile
  const int hi = lane >> 4;
  const int nc = lane & 15;
#pragma unroll
  for (int mi = 0; mi < 2; ++mi) {
#pragma unroll
    for (int t = 0; t < 4; ++t) {
      int n = nBase + t * 16 + nc;
      float bv = bias ? bias[n] : 0.f;
#pragma unroll
      for (int r = 0; r < 8; ++r) {
        int mG = mBase + wave * 32 + mi * 16 + r + 8 * hi;
        float v = acc[mi][t][r] + bv;
        if (flags & ADD_RES) v += resid[(size_t)mG * N + n];
        if (flags & OUT_TRANS)     outH[(size_t)n * ldT + mG] = (__bf16)v;
        else if (flags & OUT_BF16) outH[(size_t)mG * N + n]   = (__bf16)v;
        else                       outF[(size_t)mG * N + n]   = v;
      }
    }
  }
}

// One workgroup = (batch, 16-query tile). Q tile staged once in LDS (async),
// Q.K^T via WMMA into a 16x2048 f32 LDS strip, then softmax -> bf16 P.
__global__ void __launch_bounds__(256) attn_scores_softmax(
    const __bf16* __restrict__ Q, const __bf16* __restrict__ K,
    __bf16* __restrict__ P, float scale)
{
  extern __shared__ char smem[];
  float*  sS = (float*)smem;                              // [16][SKV_] 128 KiB
  float*  sR = (float*)(smem + 16 * SKV_ * 4);            // [16][16]
  __bf16* sQ = (__bf16*)(smem + 16 * SKV_ * 4 + 1024);    // [16][C_]   32 KiB
  const int tid  = threadIdx.x;
  const int lane = tid & 31;
  const int wave = tid >> 5;
  const int b  = blockIdx.y;
  const int qt = blockIdx.x;

  const __bf16* Q0 = Q + ((size_t)b * SQ_ + (size_t)qt * 16) * C_;
  const __bf16* K0 = K + (size_t)b * SKV_ * C_;
  const int hi = lane >> 4;
  const int nc = lane & 15;

  // stage Q tile (16x1024 contiguous) into LDS
#pragma unroll
  for (int l = 0; l < 8; ++l) {
    int e = (tid + l * 256) * 8;
    async_copy_b128(Q0 + e, sQ + e);
  }
  async_wait_all();
  __syncthreads();

  // Phase 1: each wave owns key tiles kt = wave, wave+8, ...
  for (int kt = wave; kt < SKV_ / 16; kt += 8) {
    v8f acc;
#pragma unroll
    for (int i = 0; i < 8; ++i) acc[i] = 0.f;
    for (int c0 = 0; c0 < C_; c0 += 32) {
      v16bf af  = load_frag(sQ + c0, C_);
      v16bf bfr = load_frag(K0 + (size_t)kt * 16 * C_ + c0, C_);
      acc = __builtin_amdgcn_wmma_f32_16x16x32_bf16(
          false, af, false, bfr, (short)0, acc, false, false);
    }
#pragma unroll
    for (int r = 0; r < 8; ++r) {
      int m = r + 8 * hi;
      sS[m * SKV_ + kt * 16 + nc] = acc[r] * scale;
    }
  }
  __syncthreads();

  // Phase 2: softmax, 16 threads per query row
  const int row = tid >> 4;
  const int sub = tid & 15;
  float mx = -3.4e38f;
  for (int j = sub; j < SKV_; j += 16) mx = fmaxf(mx, sS[row * SKV_ + j]);
  sR[row * 16 + sub] = mx;
  __syncthreads();
  if (sub == 0) {
    float m = sR[row * 16];
    for (int t = 1; t < 16; ++t) m = fmaxf(m, sR[row * 16 + t]);
    sR[row * 16] = m;
  }
  __syncthreads();
  mx = sR[row * 16];
  __syncthreads();
  float sm = 0.f;
  for (int j = sub; j < SKV_; j += 16) sm += __expf(sS[row * SKV_ + j] - mx);
  sR[row * 16 + sub] = sm;
  __syncthreads();
  if (sub == 0) {
    float s = 0.f;
    for (int t = 0; t < 16; ++t) s += sR[row * 16 + t];
    sR[row * 16] = s;
  }
  __syncthreads();
  const float inv = 1.f / sR[row * 16];
  __bf16* Pr = P + ((size_t)b * SQ_ + (size_t)qt * 16 + row) * SKV_;
  for (int j = sub; j < SKV_; j += 16)
    Pr[j] = (__bf16)(__expf(sS[row * SKV_ + j] - mx) * inv);
}

__global__ void __launch_bounds__(256) f32_to_bf16_k(
    const float* __restrict__ in, __bf16* __restrict__ out, long long n4)
{
  long long i = (long long)blockIdx.x * blockDim.x + threadIdx.x;
  long long stride = (long long)gridDim.x * blockDim.x;
  for (; i < n4; i += stride) {
    f32x4 v = ((const f32x4*)in)[i];
    bf16x4 o;
#pragma unroll
    for (int j = 0; j < 4; ++j) o[j] = (__bf16)v[j];
    ((bf16x4*)out)[i] = o;
  }
}

extern "C" void kernel_launch(void* const* d_in, const int* in_sizes, int n_in,
                              void* d_out, int out_size, void* d_ws, size_t ws_size,
                              hipStream_t stream) {
  const float* x  = (const float*)d_in[0];
  const float* ce = (const float*)d_in[1];
  const float* Wq = (const float*)d_in[2];
  const float* bq = (const float*)d_in[3];
  const float* Wk = (const float*)d_in[4];
  const float* bk = (const float*)d_in[5];
  const float* Wv = (const float*)d_in[6];
  const float* bv = (const float*)d_in[7];
  const float* Wo = (const float*)d_in[8];
  const float* bo = (const float*)d_in[9];
  float* out = (float*)d_out;

  // workspace layout (bf16 buffers), ~136 MB total
  char* ws = (char*)d_ws;
  const size_t SZ_XE = (size_t)B_ * SQ_ * C_ * 2;  // 16 MiB
  const size_t SZ_W  = (size_t)C_ * C_ * 2;        //  2 MiB
  __bf16* Xb  = (__bf16*)(ws);
  __bf16* Eb  = (__bf16*)(ws + SZ_XE);
  __bf16* Wqb = (__bf16*)(ws + 2 * SZ_XE);
  __bf16* Wkb = (__bf16*)(ws + 2 * SZ_XE + SZ_W);
  __bf16* Wvb = (__bf16*)(ws + 2 * SZ_XE + 2 * SZ_W);
  __bf16* Wob = (__bf16*)(ws + 2 * SZ_XE + 3 * SZ_W);
  __bf16* Qb  = (__bf16*)(ws + 2 * SZ_XE + 4 * SZ_W);
  __bf16* Kb  = (__bf16*)(ws + 3 * SZ_XE + 4 * SZ_W);
  __bf16* Vtb = (__bf16*)(ws + 4 * SZ_XE + 4 * SZ_W);  // V^T per batch [C][SKV]
  __bf16* Pb  = (__bf16*)(ws + 5 * SZ_XE + 4 * SZ_W);  // 32 MiB
  __bf16* Ub  = (__bf16*)(ws + 7 * SZ_XE + 4 * SZ_W);
  (void)in_sizes; (void)n_in; (void)out_size; (void)ws_size;

  // 1) fp32 -> bf16 conversions (vectorized x4)
  f32_to_bf16_k<<<1024, 256, 0, stream>>>(x,  Xb, (long long)B_ * SQ_ * C_ / 4);
  f32_to_bf16_k<<<1024, 256, 0, stream>>>(ce, Eb, (long long)B_ * SKV_ * C_ / 4);
  f32_to_bf16_k<<<256,  256, 0, stream>>>(Wq, Wqb, (long long)C_ * C_ / 4);
  f32_to_bf16_k<<<256,  256, 0, stream>>>(Wk, Wkb, (long long)C_ * C_ / 4);
  f32_to_bf16_k<<<256,  256, 0, stream>>>(Wv, Wvb, (long long)C_ * C_ / 4);
  f32_to_bf16_k<<<256,  256, 0, stream>>>(Wo, Wob, (long long)C_ * C_ / 4);

  // 2) Q = x Wq^T + bq ; K = ce Wk^T + bk   (batch folded into M)
  gemm_rr<<<dim3(C_ / 64, (B_ * SQ_) / 256), 256, 0, stream>>>(
      Xb, Wqb, bq, nullptr, nullptr, Qb, B_ * SQ_, C_, C_, 0, OUT_BF16);
  gemm_rr<<<dim3(C_ / 64, (B_ * SKV_) / 256), 256, 0, stream>>>(
      Eb, Wkb, bk, nullptr, nullptr, Kb, B_ * SKV_, C_, C_, 0, OUT_BF16);

  // 3) V^T = (ce Wv^T + bv)^T per batch (transposed store so P*V is row x row)
  for (int b = 0; b < B_; ++b)
    gemm_rr<<<dim3(C_ / 64, SKV_ / 256), 256, 0, stream>>>(
        Eb + (size_t)b * SKV_ * C_, Wvb, bv, nullptr, nullptr,
        Vtb + (size_t)b * C_ * SKV_, SKV_, C_, C_, SKV_, OUT_BF16 | OUT_TRANS);

  // 4) P = softmax(Q K^T / 32)  -> bf16
  attn_scores_softmax<<<dim3(SQ_ / 16, B_), 256,
                        16 * SKV_ * 4 + 1024 + 16 * C_ * 2, stream>>>(
      Qb, Kb, Pb, 1.0f / 32.0f);

  // 5) U = x + P V (per batch, fused residual, bf16)
  for (int b = 0; b < B_; ++b)
    gemm_rr<<<dim3(C_ / 64, SQ_ / 256), 256, 0, stream>>>(
        Pb + (size_t)b * SQ_ * SKV_, Vtb + (size_t)b * C_ * SKV_,
        nullptr, x + (size_t)b * SQ_ * C_, nullptr,
        Ub + (size_t)b * SQ_ * C_, SQ_, C_, SKV_, 0, OUT_BF16 | ADD_RES);

  // 6) out = U Wo^T + bo  (fp32 output)
  gemm_rr<<<dim3(C_ / 64, (B_ * SQ_) / 256), 256, 0, stream>>>(
      Ub, Wob, bo, nullptr, out, nullptr, B_ * SQ_, C_, C_, 0, 0);
}
